// Attention_9345848836379
// MI455X (gfx1250) — compile-verified, matrix operand order checked by reading, not run on hardware
//
#include <hip/hip_runtime.h>

// ---------------------------------------------------------------------------
// MI455X (gfx1250) dual-stream attention block.
// Pipeline: f32->f16 convert, QKV GEMM (WMMA, async-LDS double-buffered),
// flash attention star/hat streams (WMMA QK^T + PV, online softmax, async-LDS
// tile staging), output-proj GEMM. All matmuls: v_wmma_f32_16x16x32_f16.
// ---------------------------------------------------------------------------

typedef __attribute__((ext_vector_type(16))) _Float16 v16h;
typedef __attribute__((ext_vector_type(8)))  float    v8f;
typedef __attribute__((ext_vector_type(4)))  int      i32x4;

constexpr int Bc = 2, Tc = 1024, Cc = 1024, Hc = 16, Dc = 64;
constexpr int Nqkv  = 3 * Cc;    // 3072
constexpr int Mrows = Bc * Tc;   // 2048
constexpr float kScale  = 0.125f;   // 1/sqrt(D)
constexpr float kNegBig = -1e30f;

#if __has_builtin(__builtin_amdgcn_global_load_async_to_lds_b128) && \
    __has_builtin(__builtin_amdgcn_s_wait_asynccnt)
#define USE_ASYNC_LDS 1
#else
#define USE_ASYNC_LDS 0
#endif

// Copy 16 bytes global -> LDS. Async path: GLOBAL_LOAD_ASYNC_TO_LDS_B128,
// tracked by ASYNCcnt, no VGPR round-trip. Builtin signature (from clang
// diagnostic): (int4 addrspace(1)* src, int4 addrspace(3)* dst, imm, imm).
__device__ inline void async_copy16(const _Float16* g, _Float16* l) {
#if USE_ASYNC_LDS
  __builtin_amdgcn_global_load_async_to_lds_b128(
      (__attribute__((address_space(1))) i32x4*)g,
      (__attribute__((address_space(3))) i32x4*)l, 0, 0);
#else
  *reinterpret_cast<float4*>(l) = *reinterpret_cast<const float4*>(g);
#endif
}

__device__ inline void wait_async_leq(int n) {
#if USE_ASYNC_LDS
  if (n == 0) __builtin_amdgcn_s_wait_asynccnt(0);
  else        __builtin_amdgcn_s_wait_asynccnt(4);
#else
  (void)n;
#endif
}

__device__ inline v8f zero8() { v8f z = {}; return z; }

__device__ inline v8f wmma16(v16h a, v16h b, v8f c) {
  return __builtin_amdgcn_wmma_f32_16x16x32_f16(false, a, false, b, (short)0, c,
                                                false, false);
}

// A fragment (16x32, MxK) from LDS row-major tile p[row][k], leading dim ld.
// Lane layout (ISA 7.12.2): lanes 0-15 -> M=lane, elems 0-7 = K0-7, 8-15 =
// K16-23; lanes 16-31 -> M=lane-16, elems = K8-15 / K24-31.
__device__ inline v16h ldsA(const _Float16* p, int row_base, int k_base, int ld) {
  const int lane = threadIdx.x & 31;
  const int m = lane & 15, hs = lane >> 4;
  v16h f;
#pragma unroll
  for (int j = 0; j < 16; ++j) {
    int k = ((j >> 3) << 4) + (hs << 3) + (j & 7);
    f[j] = p[(row_base + m) * ld + k_base + k];
  }
  return f;
}

// B fragment (32x16, KxN) where LDS holds B row-major: p[k][n].
__device__ inline v16h ldsB_rm(const _Float16* p, int k_base, int n_base, int ld) {
  const int lane = threadIdx.x & 31;
  const int n = lane & 15, hs = lane >> 4;
  v16h f;
#pragma unroll
  for (int j = 0; j < 16; ++j) {
    int k = ((j >> 3) << 4) + (hs << 3) + (j & 7);
    f[j] = p[(k_base + k) * ld + n_base + n];
  }
  return f;
}

// B fragment (32x16, KxN) where LDS holds the transpose: p[n][k].
__device__ inline v16h ldsB_tr(const _Float16* p, int k_base, int n_base, int ld) {
  const int lane = threadIdx.x & 31;
  const int n = lane & 15, hs = lane >> 4;
  v16h f;
#pragma unroll
  for (int j = 0; j < 16; ++j) {
    int k = ((j >> 3) << 4) + (hs << 3) + (j & 7);
    f[j] = p[(n_base + n) * ld + k_base + k];
  }
  return f;
}

// ---------------------------------------------------------------------------
// Vectorized f32 -> f16 convert: 8 elements per thread per step, b128 I/O.
__global__ void cvt_kernel(const float* __restrict__ src,
                           _Float16* __restrict__ dst, int n8) {
  int i = blockIdx.x * blockDim.x + threadIdx.x;
  int stride = gridDim.x * blockDim.x;
  for (; i < n8; i += stride) {
    float4 a = reinterpret_cast<const float4*>(src)[i * 2];
    float4 b = reinterpret_cast<const float4*>(src)[i * 2 + 1];
    _Float16 h[8] = {(_Float16)a.x, (_Float16)a.y, (_Float16)a.z, (_Float16)a.w,
                     (_Float16)b.x, (_Float16)b.y, (_Float16)b.z, (_Float16)b.w};
    reinterpret_cast<float4*>(dst)[i] = *reinterpret_cast<float4*>(h);
  }
}

// ---------------------------------------------------------------------------
// Tiled GEMM: [2048, 1024] x [1024, N] + bias, double-buffered async-LDS.
// MODE 0: N=3072, epilogue scatters to Q/K/V in [B,H,T,D] f16 layout.
// MODE 1: N=1024, epilogue writes fp32 row-major to Out.
// 256 thr = 8 waves, tile 128x128, 8 wmma per wave per k-step.
template <int MODE>
__global__ __launch_bounds__(256) void gemm_kernel(
    const _Float16* __restrict__ A, const _Float16* __restrict__ Bw,
    const float* __restrict__ bias, _Float16* __restrict__ Qd,
    _Float16* __restrict__ Kd, _Float16* __restrict__ Vd,
    float* __restrict__ Out) {
  constexpr int N = (MODE == 0) ? Nqkv : Cc;
  constexpr int KK = Cc;
  constexpr int NIT = KK / 32;
  __shared__ _Float16 sA[2][128][40];
  __shared__ _Float16 sB[2][32][136];
  const int m0 = blockIdx.y * 128;
  const int n0 = blockIdx.x * 128;
  const int tid = threadIdx.x;
  const int lane = tid & 31;
  const int w = tid >> 5;
  const int wm = (w >> 1) * 32;   // 4 wave-rows
  const int wn = (w & 1) * 64;    // 2 wave-cols

  // Per-thread staging: 2 A chunks + 2 B chunks = 4 async instrs per stage.
  auto issue_stage = [&](int buf, int k0) {
#pragma unroll
    for (int c = 0; c < 2; ++c) {
      int chunk = tid * 2 + c;
      int row = chunk >> 2;
      int col = (chunk & 3) * 8;
      async_copy16(&A[(size_t)(m0 + row) * KK + k0 + col], &sA[buf][row][col]);
    }
#pragma unroll
    for (int c = 0; c < 2; ++c) {
      int chunk = tid * 2 + c;
      int row = chunk >> 4;
      int col = (chunk & 15) * 8;
      async_copy16(&Bw[(size_t)(k0 + row) * N + n0 + col], &sB[buf][row][col]);
    }
  };

  v8f acc[2][4];
#pragma unroll
  for (int i = 0; i < 2; ++i)
#pragma unroll
    for (int j = 0; j < 4; ++j) acc[i][j] = zero8();

  issue_stage(0, 0);
  for (int it = 0; it < NIT; ++it) {
    const int buf = it & 1;
    if (it + 1 < NIT) {
      issue_stage(buf ^ 1, (it + 1) * 32);
      wait_async_leq(4);   // stage `it` landed; stage it+1 still in flight
    } else {
      wait_async_leq(0);
    }
    __syncthreads();

    v16h af[2], bf[4];
#pragma unroll
    for (int mt = 0; mt < 2; ++mt)
      af[mt] = ldsA(&sA[buf][0][0], wm + mt * 16, 0, 40);
#pragma unroll
    for (int nt = 0; nt < 4; ++nt)
      bf[nt] = ldsB_rm(&sB[buf][0][0], 0, wn + nt * 16, 136);
#pragma unroll
    for (int mt = 0; mt < 2; ++mt)
#pragma unroll
      for (int nt = 0; nt < 4; ++nt)
        acc[mt][nt] = wmma16(af[mt], bf[nt], acc[mt][nt]);
    __syncthreads();   // everyone done with buf before it is refilled
  }

  // Epilogue. C/D layout: lane l elem e -> row (l/16)*8+e, col l%16.
#pragma unroll
  for (int mt = 0; mt < 2; ++mt)
#pragma unroll
    for (int nt = 0; nt < 4; ++nt) {
      v8f a = acc[mt][nt];
#pragma unroll
      for (int e = 0; e < 8; ++e) {
        int row = m0 + wm + mt * 16 + ((lane >> 4) << 3) + e;
        int col = n0 + wn + nt * 16 + (lane & 15);
        float val = a[e] + bias[col];
        if (MODE == 0) {
          int s = col >> 10, cc = col & (Cc - 1);
          int h = cc >> 6, d = cc & (Dc - 1);
          int b = row >> 10, t = row & (Tc - 1);
          _Float16* dst = (s == 0) ? Qd : (s == 1) ? Kd : Vd;
          dst[((size_t)(b * Hc + h) * Tc + t) * Dc + d] = (_Float16)val;
        } else {
          Out[(size_t)row * Cc + col] = val;
        }
      }
    }
}

// ---------------------------------------------------------------------------
// Flash causal attention (star stream). Grid (T/64, B*H), 128 thr = 4 waves,
// each wave owns 16 query rows. Online softmax in f32 registers.
__global__ __launch_bounds__(128) void attn_star_kernel(
    const _Float16* __restrict__ Q, const _Float16* __restrict__ K,
    const _Float16* __restrict__ V, _Float16* __restrict__ Y) {
  __shared__ _Float16 sQ[64][72], sK[64][72], sV[64][72];
  __shared__ _Float16 sP[4][16][72];
  const int bh = blockIdx.y;
  const int qb = blockIdx.x;
  const int q0 = qb * 64;
  const int tid = threadIdx.x;
  const int lane = tid & 31;
  const int w = tid >> 5;
  const size_t base = (size_t)bh * Tc * Dc;

#pragma unroll
  for (int c = 0; c < 4; ++c) {
    int chunk = tid * 4 + c;
    int row = chunk >> 3;
    int col = (chunk & 7) * 8;
    async_copy16(&Q[base + (size_t)(q0 + row) * Dc + col], &sQ[row][col]);
  }
  wait_async_leq(0);
  __syncthreads();
  v16h aq0 = ldsA(&sQ[0][0], w * 16, 0, 72);
  v16h aq1 = ldsA(&sQ[0][0], w * 16, 32, 72);

  float mrow[8], lrow[8];
  v8f O[4];
#pragma unroll
  for (int e = 0; e < 8; ++e) { mrow[e] = kNegBig; lrow[e] = 0.f; }
#pragma unroll
  for (int nt = 0; nt < 4; ++nt) O[nt] = zero8();

  for (int kb = 0; kb <= qb; ++kb) {
    const int kbase = kb * 64;
#pragma unroll
    for (int c = 0; c < 4; ++c) {
      int chunk = tid * 4 + c;
      int row = chunk >> 3;
      int col = (chunk & 7) * 8;
      async_copy16(&K[base + (size_t)(kbase + row) * Dc + col], &sK[row][col]);
      async_copy16(&V[base + (size_t)(kbase + row) * Dc + col], &sV[row][col]);
    }
    if (kb + 1 <= qb) {  // global_prefetch_b8 of next key block
      __builtin_prefetch(&K[base + (size_t)(kbase + 64 + (tid >> 1)) * Dc], 0, 1);
      __builtin_prefetch(&V[base + (size_t)(kbase + 64 + (tid >> 1)) * Dc], 0, 1);
    }
    wait_async_leq(0);
    __syncthreads();

    v8f s[4];
#pragma unroll
    for (int nt = 0; nt < 4; ++nt) {
      v8f t0 = zero8();
      t0 = wmma16(aq0, ldsB_tr(&sK[0][0], 0, nt * 16, 72), t0);
      t0 = wmma16(aq1, ldsB_tr(&sK[0][0], 32, nt * 16, 72), t0);
      s[nt] = t0;
    }
#pragma unroll
    for (int nt = 0; nt < 4; ++nt)
#pragma unroll
      for (int e = 0; e < 8; ++e) {
        int qrow = q0 + w * 16 + ((lane >> 4) << 3) + e;
        int key = kbase + nt * 16 + (lane & 15);
        s[nt][e] = (key > qrow) ? kNegBig : s[nt][e] * kScale;
      }
#pragma unroll
    for (int e = 0; e < 8; ++e) {
      float mx = fmaxf(fmaxf(s[0][e], s[1][e]), fmaxf(s[2][e], s[3][e]));
#pragma unroll
      for (int off = 1; off < 16; off <<= 1) mx = fmaxf(mx, __shfl_xor(mx, off, 32));
      float mn = fmaxf(mrow[e], mx);
      float alpha = __expf(mrow[e] - mn);
      mrow[e] = mn;
      float sum = 0.f;
#pragma unroll
      for (int nt = 0; nt < 4; ++nt) {
        float p = __expf(s[nt][e] - mn);
        s[nt][e] = p;
        sum += p;
        O[nt][e] *= alpha;
      }
#pragma unroll
      for (int off = 1; off < 16; off <<= 1) sum += __shfl_xor(sum, off, 32);
      lrow[e] = lrow[e] * alpha + sum;
    }
    // Repack P (D-layout) -> A-layout via per-wave LDS tile (wave-local,
    // DS ops are in order within a wave; no barrier needed).
#pragma unroll
    for (int nt = 0; nt < 4; ++nt)
#pragma unroll
      for (int e = 0; e < 8; ++e) {
        int r = ((lane >> 4) << 3) + e;
        int kl = nt * 16 + (lane & 15);
        sP[w][r][kl] = (_Float16)s[nt][e];
      }
    v16h ap0 = ldsA(&sP[w][0][0], 0, 0, 72);
    v16h ap1 = ldsA(&sP[w][0][0], 0, 32, 72);
#pragma unroll
    for (int nt = 0; nt < 4; ++nt) {
      O[nt] = wmma16(ap0, ldsB_rm(&sV[0][0], 0, nt * 16, 72), O[nt]);
      O[nt] = wmma16(ap1, ldsB_rm(&sV[0][0], 32, nt * 16, 72), O[nt]);
    }
    __syncthreads();
  }

  const int b = bh >> 4, h = bh & 15;
#pragma unroll
  for (int nt = 0; nt < 4; ++nt)
#pragma unroll
    for (int e = 0; e < 8; ++e) {
      int qidx = q0 + w * 16 + ((lane >> 4) << 3) + e;
      int d = nt * 16 + (lane & 15);
      float val = O[nt][e] / lrow[e];
      Y[((size_t)(b * Tc + qidx)) * Cc + h * Dc + d] = (_Float16)val;
    }
}

// ---------------------------------------------------------------------------
// Hat stream: merged scores (q_h.k_s causal, q_h.k_h on diagonal), output
// sum_{j<=i} p_j v_s[j] + p_i v_h[i]. Diagonal extra term done via a second
// WMMA pass with off-diagonal P zeroed in LDS.
__global__ __launch_bounds__(128) void attn_hat_kernel(
    const _Float16* __restrict__ Qh, const _Float16* __restrict__ Ks,
    const _Float16* __restrict__ Kh, const _Float16* __restrict__ Vs,
    const _Float16* __restrict__ Vh, _Float16* __restrict__ Y) {
  __shared__ _Float16 sQ[64][72], sK[64][72], sV[64][72];
  __shared__ _Float16 sKh[64][72], sVh[64][72];
  __shared__ _Float16 sP[4][16][72];
  const int bh = blockIdx.y;
  const int qb = blockIdx.x;
  const int q0 = qb * 64;
  const int tid = threadIdx.x;
  const int lane = tid & 31;
  const int w = tid >> 5;
  const size_t base = (size_t)bh * Tc * Dc;

#pragma unroll
  for (int c = 0; c < 4; ++c) {
    int chunk = tid * 4 + c;
    int row = chunk >> 3;
    int col = (chunk & 7) * 8;
    async_copy16(&Qh[base + (size_t)(q0 + row) * Dc + col], &sQ[row][col]);
  }
  wait_async_leq(0);
  __syncthreads();
  v16h aq0 = ldsA(&sQ[0][0], w * 16, 0, 72);
  v16h aq1 = ldsA(&sQ[0][0], w * 16, 32, 72);

  float mrow[8], lrow[8];
  v8f O[4];
#pragma unroll
  for (int e = 0; e < 8; ++e) { mrow[e] = kNegBig; lrow[e] = 0.f; }
#pragma unroll
  for (int nt = 0; nt < 4; ++nt) O[nt] = zero8();

  for (int kb = 0; kb <= qb; ++kb) {
    const int kbase = kb * 64;
    const bool diag = (kb == qb);
#pragma unroll
    for (int c = 0; c < 4; ++c) {
      int chunk = tid * 4 + c;
      int row = chunk >> 3;
      int col = (chunk & 7) * 8;
      async_copy16(&Ks[base + (size_t)(kbase + row) * Dc + col], &sK[row][col]);
      async_copy16(&Vs[base + (size_t)(kbase + row) * Dc + col], &sV[row][col]);
    }
    if (diag) {
#pragma unroll
      for (int c = 0; c < 4; ++c) {
        int chunk = tid * 4 + c;
        int row = chunk >> 3;
        int col = (chunk & 7) * 8;
        async_copy16(&Kh[base + (size_t)(kbase + row) * Dc + col], &sKh[row][col]);
        async_copy16(&Vh[base + (size_t)(kbase + row) * Dc + col], &sVh[row][col]);
      }
    }
    wait_async_leq(0);
    __syncthreads();

    v8f s[4];
#pragma unroll
    for (int nt = 0; nt < 4; ++nt) {
      v8f t0 = zero8();
      t0 = wmma16(aq0, ldsB_tr(&sK[0][0], 0, nt * 16, 72), t0);
      t0 = wmma16(aq1, ldsB_tr(&sK[0][0], 32, nt * 16, 72), t0);
      s[nt] = t0;
    }
    if (diag) {
      v8f sh[4];
#pragma unroll
      for (int nt = 0; nt < 4; ++nt) {
        v8f t0 = zero8();
        t0 = wmma16(aq0, ldsB_tr(&sKh[0][0], 0, nt * 16, 72), t0);
        t0 = wmma16(aq1, ldsB_tr(&sKh[0][0], 32, nt * 16, 72), t0);
        sh[nt] = t0;
      }
#pragma unroll
      for (int nt = 0; nt < 4; ++nt)
#pragma unroll
        for (int e = 0; e < 8; ++e) {
          int qrow = q0 + w * 16 + ((lane >> 4) << 3) + e;
          int key = kbase + nt * 16 + (lane & 15);
          float v = (key == qrow) ? sh[nt][e] : s[nt][e];
          s[nt][e] = (key > qrow) ? kNegBig : v * kScale;
        }
    } else {
#pragma unroll
      for (int nt = 0; nt < 4; ++nt)
#pragma unroll
        for (int e = 0; e < 8; ++e) s[nt][e] *= kScale;
    }
#pragma unroll
    for (int e = 0; e < 8; ++e) {
      float mx = fmaxf(fmaxf(s[0][e], s[1][e]), fmaxf(s[2][e], s[3][e]));
#pragma unroll
      for (int off = 1; off < 16; off <<= 1) mx = fmaxf(mx, __shfl_xor(mx, off, 32));
      float mn = fmaxf(mrow[e], mx);
      float alpha = __expf(mrow[e] - mn);
      mrow[e] = mn;
      float sum = 0.f;
#pragma unroll
      for (int nt = 0; nt < 4; ++nt) {
        float p = __expf(s[nt][e] - mn);
        s[nt][e] = p;
        sum += p;
        O[nt][e] *= alpha;
      }
#pragma unroll
      for (int off = 1; off < 16; off <<= 1) sum += __shfl_xor(sum, off, 32);
      lrow[e] = lrow[e] * alpha + sum;
    }
#pragma unroll
    for (int nt = 0; nt < 4; ++nt)
#pragma unroll
      for (int e = 0; e < 8; ++e) {
        int r = ((lane >> 4) << 3) + e;
        int kl = nt * 16 + (lane & 15);
        sP[w][r][kl] = (_Float16)s[nt][e];
      }
    v16h ap0 = ldsA(&sP[w][0][0], 0, 0, 72);
    v16h ap1 = ldsA(&sP[w][0][0], 0, 32, 72);
#pragma unroll
    for (int nt = 0; nt < 4; ++nt) {
      O[nt] = wmma16(ap0, ldsB_rm(&sV[0][0], 0, nt * 16, 72), O[nt]);
      O[nt] = wmma16(ap1, ldsB_rm(&sV[0][0], 32, nt * 16, 72), O[nt]);
    }
    if (diag) {
      // Keep only diagonal probabilities, accumulate p_ii * v_hat[i].
#pragma unroll
      for (int nt = 0; nt < 4; ++nt)
#pragma unroll
        for (int e = 0; e < 8; ++e) {
          int r = ((lane >> 4) << 3) + e;
          int kl = nt * 16 + (lane & 15);
          int qrow = q0 + w * 16 + r;
          int key = kbase + kl;
          if (key != qrow) sP[w][r][kl] = (_Float16)0.0f;
        }
      v16h ad0 = ldsA(&sP[w][0][0], 0, 0, 72);
      v16h ad1 = ldsA(&sP[w][0][0], 0, 32, 72);
#pragma unroll
      for (int nt = 0; nt < 4; ++nt) {
        O[nt] = wmma16(ad0, ldsB_rm(&sVh[0][0], 0, nt * 16, 72), O[nt]);
        O[nt] = wmma16(ad1, ldsB_rm(&sVh[0][0], 32, nt * 16, 72), O[nt]);
      }
    }
    __syncthreads();
  }

  const int b = bh >> 4, h = bh & 15;
#pragma unroll
  for (int nt = 0; nt < 4; ++nt)
#pragma unroll
    for (int e = 0; e < 8; ++e) {
      int qidx = q0 + w * 16 + ((lane >> 4) << 3) + e;
      int d = nt * 16 + (lane & 15);
      float val = O[nt][e] / lrow[e];
      Y[((size_t)(b * Tc + qidx)) * Cc + h * Dc + d] = (_Float16)val;
    }
}

// ---------------------------------------------------------------------------
extern "C" void kernel_launch(void* const* d_in, const int* in_sizes, int n_in,
                              void* d_out, int out_size, void* d_ws,
                              size_t ws_size, hipStream_t stream) {
  (void)in_sizes; (void)n_in; (void)out_size; (void)ws_size;
  const float* x_star = (const float*)d_in[0];
  const float* x_hat  = (const float*)d_in[1];
  // d_in[2], d_in[3]: keep masks — structurally fixed (causal / diag), unused.
  const float* W_attn = (const float*)d_in[4];
  const float* b_attn = (const float*)d_in[5];
  const float* W_proj = (const float*)d_in[6];
  const float* b_proj = (const float*)d_in[7];

  const size_t nx = (size_t)Bc * Tc * Cc;  // 2,097,152 (== B*H*T*D)
  _Float16* ws = (_Float16*)d_ws;
  _Float16* xs  = ws; ws += nx;
  _Float16* xh  = ws; ws += nx;
  _Float16* Wa  = ws; ws += (size_t)Cc * Nqkv;
  _Float16* Wp  = ws; ws += (size_t)Cc * Cc;
  _Float16* Qs  = ws; ws += nx;   // [B,H,T,D]
  _Float16* KsD = ws; ws += nx;
  _Float16* VsD = ws; ws += nx;
  _Float16* Qhd = ws; ws += nx;
  _Float16* KhD = ws; ws += nx;
  _Float16* VhD = ws; ws += nx;
  _Float16* Ys  = ws; ws += nx;   // [B,T,C]
  _Float16* Yh  = ws; ws += nx;

  cvt_kernel<<<dim3(512), dim3(256), 0, stream>>>(x_star, xs, (int)(nx / 8));
  cvt_kernel<<<dim3(512), dim3(256), 0, stream>>>(x_hat, xh, (int)(nx / 8));
  cvt_kernel<<<dim3(512), dim3(256), 0, stream>>>(W_attn, Wa, Cc * Nqkv / 8);
  cvt_kernel<<<dim3(512), dim3(256), 0, stream>>>(W_proj, Wp, Cc * Cc / 8);

  dim3 gq(Nqkv / 128, Mrows / 128);
  gemm_kernel<0><<<gq, dim3(256), 0, stream>>>(xs, Wa, b_attn, Qs, KsD, VsD, nullptr);
  gemm_kernel<0><<<gq, dim3(256), 0, stream>>>(xh, Wa, b_attn, Qhd, KhD, VhD, nullptr);

  dim3 ga(Tc / 64, Bc * Hc);
  attn_star_kernel<<<ga, dim3(128), 0, stream>>>(Qs, KsD, VsD, Ys);
  attn_hat_kernel<<<ga, dim3(128), 0, stream>>>(Qhd, KsD, KhD, VsD, VhD, Yh);

  float* out = (float*)d_out;
  dim3 gp(Cc / 128, Mrows / 128);
  gemm_kernel<1><<<gp, dim3(256), 0, stream>>>(Ys, Wp, b_proj, nullptr, nullptr, nullptr, out);
  gemm_kernel<1><<<gp, dim3(256), 0, stream>>>(Yh, Wp, b_proj, nullptr, nullptr, nullptr, out + nx);
}
